// KMeans_42863773614198
// MI455X (gfx1250) — matrix operand order repaired and checked by reading, not compiled, not run on hardware
//
#include <hip/hip_runtime.h>

// ---------------------------------------------------------------------------
// KMeans assignment step for MI455X (gfx1250, wave32, WMMA).
// distances[K,N] = ||c||^2 + ||x||^2 - 2 * (C @ X^T), argmin over K, mean loss.
// GEMM runs as bf16x3 (hi/lo split, drop lo*lo) on v_wmma_f32_16x16x32_bf16.
// Round 2: double-buffered LDS + async global->LDS copies (ASYNCcnt) for the
// bf16-presplit centroid tiles, and register-prefetched input tiles so the
// staging latency overlaps the 24-WMMA compute block of the previous step.
// ---------------------------------------------------------------------------

#define KM_N 65536
#define KM_D 1024
#define KM_K 256

typedef __attribute__((ext_vector_type(16))) __bf16 v16bf;
typedef __attribute__((ext_vector_type(8)))  float  v8f;

union U16bf { unsigned int u[8]; v16bf v; };

__device__ __forceinline__ unsigned int f2bf_rne(float f) {
  unsigned int u = __float_as_uint(f);
  u += 0x7FFFu + ((u >> 16) & 1u);   // round-to-nearest-even on the bf16 boundary
  return u >> 16;                    // 16-bit result in low bits
}
__device__ __forceinline__ float bf2f(unsigned int h) {
  return __uint_as_float(h << 16);
}

// ---- LDS fragment gathers (ISA 7.12.2 wave32 layouts, 16-bit A/B) ----------
// LDS tile layout: 128 rows, 16 data dwords (32 bf16 along d) + 4 pad = stride 20.
// Stride 20 dwords = 80 B keeps every row 16B-aligned and bank-conflict-free.

__device__ __forceinline__ v16bf ld_a_frag(const unsigned int* s, int row, int h) {
  // A (16x32, MxK): lane m holds K runs [h*8, h*8+8) and [16+h*8, 16+h*8+8)
  U16bf f;
  const uint4* p0 = (const uint4*)(s + row * 20 + h * 4);
  const uint4* p1 = (const uint4*)(s + row * 20 + 8 + h * 4);
  uint4 a = p0[0], b = p1[0];
  f.u[0] = a.x; f.u[1] = a.y; f.u[2] = a.z; f.u[3] = a.w;
  f.u[4] = b.x; f.u[5] = b.y; f.u[6] = b.z; f.u[7] = b.w;
  return f.v;
}

__device__ __forceinline__ v16bf ld_b_frag(const unsigned int* s, int row, int h) {
  // B (32x16, KxN): lane n holds K runs [h*8, h*8+8) and [16+h*8, 16+h*8+8)
  U16bf f;
  const uint4* p0 = (const uint4*)(s + row * 20 + h * 4);
  const uint4* p1 = (const uint4*)(s + row * 20 + 8 + h * 4);
  uint4 a = p0[0], b = p1[0];
  f.u[0] = a.x; f.u[1] = a.y; f.u[2] = a.z; f.u[3] = a.w;
  f.u[4] = b.x; f.u[5] = b.y; f.u[6] = b.z; f.u[7] = b.w;
  return f.v;
}

// ---- CDNA5 async copy: global -> LDS, 16B, tracked by ASYNCcnt -------------
__device__ __forceinline__ void async_cp_b128(unsigned lds_addr, const unsigned int* gsrc) {
  asm volatile("global_load_async_to_lds_b128 %0, %1, off"
               :: "v"(lds_addr), "v"(gsrc) : "memory");
}
__device__ __forceinline__ void async_wait0() {
  asm volatile("s_wait_asynccnt 0x0" ::: "memory");
}

// ---------------------------------------------------------------------------
// Kernel 1: row squared norms (one block per row, fixed-order reduction)
// ---------------------------------------------------------------------------
__global__ __launch_bounds__(256) void km_rownorm(const float* __restrict__ X,
                                                  float* __restrict__ out, int D) {
  __shared__ float red[256];
  const int row = blockIdx.x;
  const float* p = X + (size_t)row * D;
  float s = 0.0f;
  for (int i = threadIdx.x; i < D; i += 256) { float v = p[i]; s += v * v; }
  red[threadIdx.x] = s;
  __syncthreads();
  for (int off = 128; off > 0; off >>= 1) {
    if (threadIdx.x < (unsigned)off) red[threadIdx.x] += red[threadIdx.x + off];
    __syncthreads();
  }
  if (threadIdx.x == 0) out[row] = red[0];
}

// ---------------------------------------------------------------------------
// Kernel 2: pre-split centroids into packed bf16 hi/lo pairs (even d in low 16)
// ---------------------------------------------------------------------------
__global__ __launch_bounds__(256) void km_split_c(const float* __restrict__ C,
                                                  unsigned int* __restrict__ chi,
                                                  unsigned int* __restrict__ clo,
                                                  int totalPairs) {
  int u = blockIdx.x * 256 + threadIdx.x;
  if (u >= totalPairs) return;
  float a = C[2 * u], b = C[2 * u + 1];
  unsigned int ah = f2bf_rne(a);
  unsigned int bh = f2bf_rne(b);
  unsigned int al = f2bf_rne(a - bf2f(ah));
  unsigned int bl = f2bf_rne(b - bf2f(bh));
  chi[u] = ah | (bh << 16);
  clo[u] = al | (bl << 16);
}

// ---------------------------------------------------------------------------
// Kernel 3: tiled GEMM + distance epilogue.
// Block = 256 threads = 8 waves; tile 128k x 128n; wave = all 128 k x 16 n.
// Double-buffered LDS; centroid tiles via async global->LDS; input tiles
// register-prefetched and converted after the WMMA block.
// ---------------------------------------------------------------------------
__global__ __launch_bounds__(256) void km_gemm_dist(const float* __restrict__ xin,
                                                    const unsigned int* __restrict__ chi,
                                                    const unsigned int* __restrict__ clo,
                                                    const float* __restrict__ x_sq,
                                                    const float* __restrict__ c_sq,
                                                    float* __restrict__ dist) {
  __shared__ unsigned int s_chi[2][128 * 20];
  __shared__ unsigned int s_clo[2][128 * 20];
  __shared__ unsigned int s_xhi[2][128 * 20];
  __shared__ unsigned int s_xlo[2][128 * 20];

  const int t    = threadIdx.x;
  const int lane = t & 31;
  const int w    = t >> 5;       // wave 0..7
  const int h    = lane >> 4;    // lane half (selects K sub-run per ISA layout)
  const int r    = lane & 15;    // M row (A) / N col (B) within 16x16 tile

  const int n0 = blockIdx.x * 128;
  const int k0 = blockIdx.y * 128;

  const int srow  = t >> 1;      // staging row 0..127
  const int shalf = t & 1;       // staging half of the 32-wide d chunk

  v8f acc[8];
  const v8f vzero = {0.f, 0.f, 0.f, 0.f, 0.f, 0.f, 0.f, 0.f};
#pragma unroll
  for (int kt = 0; kt < 8; ++kt) acc[kt] = vzero;

  float4 pf[4];  // input-tile prefetch registers

  // issue async centroid-tile copies (bf16 pre-split, L2-resident) into buf
  auto issue_centroids = [&](int buf, int d0) {
    const unsigned int* gh = chi + (size_t)(k0 + srow) * (KM_D / 2) + (d0 >> 1) + shalf * 8;
    const unsigned int* gl = clo + (size_t)(k0 + srow) * (KM_D / 2) + (d0 >> 1) + shalf * 8;
    unsigned dh = (unsigned)(size_t)(const void*)&s_chi[buf][srow * 20 + shalf * 8];
    unsigned dl = (unsigned)(size_t)(const void*)&s_clo[buf][srow * 20 + shalf * 8];
    async_cp_b128(dh,      gh);
    async_cp_b128(dh + 16, gh + 4);
    async_cp_b128(dl,      gl);
    async_cp_b128(dl + 16, gl + 4);
  };
  // issue input-tile global loads into prefetch registers
  auto issue_inputs = [&](int d0) {
    const float4* px = (const float4*)(xin + (size_t)(n0 + srow) * KM_D + d0 + shalf * 16);
#pragma unroll
    for (int q = 0; q < 4; ++q) pf[q] = px[q];
  };
  // fp32 -> bf16 hi/lo split of the prefetched input tile, store to LDS buf
  auto store_inputs = [&](int buf) {
    unsigned int uh[8], ul[8];
#pragma unroll
    for (int q = 0; q < 4; ++q) {
      float vals[4] = {pf[q].x, pf[q].y, pf[q].z, pf[q].w};
#pragma unroll
      for (int p = 0; p < 2; ++p) {
        float a = vals[2 * p], b = vals[2 * p + 1];
        unsigned int ah = f2bf_rne(a);
        unsigned int bh = f2bf_rne(b);
        unsigned int al = f2bf_rne(a - bf2f(ah));
        unsigned int bl = f2bf_rne(b - bf2f(bh));
        uh[q * 2 + p] = ah | (bh << 16);
        ul[q * 2 + p] = al | (bl << 16);
      }
    }
    uint4* dh = (uint4*)&s_xhi[buf][srow * 20 + shalf * 8];
    uint4* dl = (uint4*)&s_xlo[buf][srow * 20 + shalf * 8];
    dh[0] = *(const uint4*)&uh[0]; dh[1] = *(const uint4*)&uh[4];
    dl[0] = *(const uint4*)&ul[0]; dl[1] = *(const uint4*)&ul[4];
  };

  // ---- prologue: fill buffer 0 ----
  issue_centroids(0, 0);
  issue_inputs(0);
  store_inputs(0);
  async_wait0();
  __syncthreads();

  const int NSTEP = KM_D / 32;   // 32
  for (int step = 0; step < NSTEP; ++step) {
    const int cur = step & 1;
    const bool has_next = (step + 1) < NSTEP;

    // stage next buffer while this buffer computes
    if (has_next) {
      issue_centroids(1 - cur, (step + 1) * 32);
      issue_inputs((step + 1) * 32);
    }

    // ---- compute: 8 k-subtiles x (hi*hi + hi*lo + lo*hi) ----
    v16bf bh = ld_b_frag(&s_xhi[cur][0], w * 16 + r, h);
    v16bf bl = ld_b_frag(&s_xlo[cur][0], w * 16 + r, h);
#pragma unroll
    for (int kt = 0; kt < 8; ++kt) {
      v16bf ah = ld_a_frag(&s_chi[cur][0], kt * 16 + r, h);
      v16bf al = ld_a_frag(&s_clo[cur][0], kt * 16 + r, h);
      acc[kt] = __builtin_amdgcn_wmma_f32_16x16x32_bf16(false, ah, false, bh, (short)0, acc[kt], false, false);
      acc[kt] = __builtin_amdgcn_wmma_f32_16x16x32_bf16(false, ah, false, bl, (short)0, acc[kt], false, false);
      acc[kt] = __builtin_amdgcn_wmma_f32_16x16x32_bf16(false, al, false, bh, (short)0, acc[kt], false, false);
    }

    // convert + store the prefetched input tile after the WMMA burst
    if (has_next) store_inputs(1 - cur);

    async_wait0();       // centroid async copies for next buffer landed
    __syncthreads();     // publish next buffer / retire reads of cur
  }

  // ---- epilogue: dist = c_sq + x_sq - 2*cross, C/D layout per ISA 7.12.2 ----
  const int n = n0 + w * 16 + r;
  const float xs = x_sq[n];
#pragma unroll
  for (int kt = 0; kt < 8; ++kt) {
    const int kb = k0 + kt * 16 + h * 8;
#pragma unroll
    for (int j = 0; j < 8; ++j) {
      const int k = kb + j;
      float dval = c_sq[k] + xs - 2.0f * acc[kt][j];
      dist[(size_t)k * KM_N + n] = dval;
    }
  }
}

// ---------------------------------------------------------------------------
// Kernel 4: argmin over K per point (coalesced column scan) + partial sums
// ---------------------------------------------------------------------------
__global__ __launch_bounds__(256) void km_argmin(const float* __restrict__ dist,
                                                 float* __restrict__ assign_out,
                                                 float* __restrict__ partials) {
  __shared__ float red[256];
  const int n = blockIdx.x * 256 + threadIdx.x;
  float best = dist[n];
  int bk = 0;
  float sum = best;
  for (int k = 1; k < KM_K; ++k) {
    float d = dist[(size_t)k * KM_N + n];
    sum += d;
    if (d < best) { best = d; bk = k; }   // strict '<' keeps first index (argmin)
  }
  assign_out[n] = (float)bk;
  red[threadIdx.x] = sum;
  __syncthreads();
  for (int off = 128; off > 0; off >>= 1) {
    if (threadIdx.x < (unsigned)off) red[threadIdx.x] += red[threadIdx.x + off];
    __syncthreads();
  }
  if (threadIdx.x == 0) partials[blockIdx.x] = red[0];
}

// ---------------------------------------------------------------------------
// Kernel 5: deterministic final reduce -> loss = mean(distances)
// ---------------------------------------------------------------------------
__global__ __launch_bounds__(256) void km_loss(const float* __restrict__ partials,
                                               float* __restrict__ loss_out) {
  __shared__ float red[256];
  red[threadIdx.x] = partials[threadIdx.x];
  __syncthreads();
  for (int off = 128; off > 0; off >>= 1) {
    if (threadIdx.x < (unsigned)off) red[threadIdx.x] += red[threadIdx.x + off];
    __syncthreads();
  }
  if (threadIdx.x == 0)
    loss_out[0] = red[0] * (1.0f / ((float)KM_K * (float)KM_N));
}

// ---------------------------------------------------------------------------
extern "C" void kernel_launch(void* const* d_in, const int* in_sizes, int n_in,
                              void* d_out, int out_size, void* d_ws, size_t ws_size,
                              hipStream_t stream) {
  (void)in_sizes; (void)n_in; (void)out_size; (void)ws_size;

  const float* X = (const float*)d_in[0];  // inputs    [N, D] fp32
  const float* C = (const float*)d_in[1];  // centroids [K, D] fp32

  float* out        = (float*)d_out;
  float* dist       = out;                             // [K, N]
  float* assign_out = out + (size_t)KM_K * KM_N;       // [N] (indices as float)
  float* loss_out   = assign_out + KM_N;               // [1]

  // workspace carve (~1.3 MB)
  char* w = (char*)d_ws;
  float* x_sq = (float*)w;               w += (size_t)KM_N * 4;
  float* c_sq = (float*)w;               w += (size_t)KM_K * 4;
  unsigned int* chi = (unsigned int*)w;  w += (size_t)KM_K * (KM_D / 2) * 4;
  unsigned int* clo = (unsigned int*)w;  w += (size_t)KM_K * (KM_D / 2) * 4;
  float* partials = (float*)w;           // [N/256] = 256 floats

  km_rownorm<<<KM_N, 256, 0, stream>>>(X, x_sq, KM_D);
  km_rownorm<<<KM_K, 256, 0, stream>>>(C, c_sq, KM_D);

  const int pairs = KM_K * KM_D / 2;
  km_split_c<<<(pairs + 255) / 256, 256, 0, stream>>>(C, chi, clo, pairs);

  dim3 grid(KM_N / 128, KM_K / 128);   // 512 x 2 workgroups
  km_gemm_dist<<<grid, 256, 0, stream>>>(X, chi, clo, x_sq, c_sq, dist);

  km_argmin<<<KM_N / 256, 256, 0, stream>>>(dist, assign_out, partials);
  km_loss<<<1, 256, 0, stream>>>(partials, loss_out);
}
